// CausalAttention_15410342658770
// MI455X (gfx1250) — matrix operand order repaired and checked by reading, not compiled, not run on hardware
//
#include <hip/hip_runtime.h>
#include <math.h>

// Problem constants (B, T, D from the reference)
#define B_  4
#define T_  2048
#define D_  1024
#define BT_ (B_*T_)

// GEMM tiling: 128x128 block tile, K-tile 32, 256 threads = 8 wave32s.
// Wave grid 4x2 -> each wave owns a 32x64 sub-tile = 2x4 WMMA 16x16 tiles.
#define BM 128
#define BN 128
#define BK 32
#define LDSK 40   // padded row stride in bf16 elements (80 B, 16B-aligned rows)

typedef __attribute__((ext_vector_type(16))) __bf16         v16bf;
typedef __attribute__((ext_vector_type(8)))  float          v8f;
typedef __attribute__((ext_vector_type(4)))  unsigned int   u32x4;
typedef __attribute__((ext_vector_type(4)))  int            i32x4;
typedef __attribute__((ext_vector_type(4)))  float          f32x4;
typedef __attribute__((ext_vector_type(4)))  unsigned short u16x4;

// Address-space-qualified pointee types for the async builtin:
// HIP-clang prints addrspace(1) as "__device__" -> param is int4 addrspace(1)*.
typedef __attribute__((address_space(1))) i32x4 gi32x4;   // global int4
typedef __attribute__((address_space(3))) i32x4 li32x4;   // LDS int4

// ---- CDNA5 async global->LDS copy (ASYNCcnt-tracked), with safe fallback ----
#if defined(__has_builtin)
#  if __has_builtin(__builtin_amdgcn_global_load_async_to_lds_b128)
#    define HAVE_ASYNC_LDS 1
#  endif
#endif

#ifdef HAVE_ASYNC_LDS
#  if __has_builtin(__builtin_amdgcn_s_wait_asynccnt)
#    define ASYNC_WAIT0() __builtin_amdgcn_s_wait_asynccnt(0)
#  else
#    define ASYNC_WAIT0() asm volatile("s_wait_asynccnt 0x0" ::: "memory")
#  endif
#else
#  define ASYNC_WAIT0() ((void)0)
#endif

__device__ __forceinline__ void copy16_g2l(const unsigned short* g, unsigned short* l) {
#ifdef HAVE_ASYNC_LDS
  __builtin_amdgcn_global_load_async_to_lds_b128((gi32x4*)g, (li32x4*)l, 0, 0);
#else
  u16x4 a = *(const u16x4*)g;
  u16x4 b = *(const u16x4*)(g + 4);
  *(u16x4*)l       = a;
  *(u16x4*)(l + 4) = b;
#endif
}

__device__ __forceinline__ unsigned short f32_to_bf16(float f) {
  union { float f; unsigned int u; } x; x.f = f;
  unsigned int lsb = (x.u >> 16) & 1u;
  return (unsigned short)((x.u + 0x7fffu + lsb) >> 16);
}

// ---- WMMA fragment loaders (bf16 16x16x32, wave32 layouts per ISA 7.12.2) ----
// A 16x32: lane L holds row M=L%16; half=L/16 selects K groups {half*8..} and {16+half*8..}
__device__ __forceinline__ v16bf load_frag_a(const unsigned short* As, int rowBase, int lane) {
  const int row  = rowBase + (lane & 15);
  const int half = (lane >> 4) << 3;                 // 0 or 8 elements
  union { v16bf v; u32x4 q[2]; } f;
  f.q[0] = *(const u32x4*)(As + row * LDSK + half);        // K = half*8   + 0..7
  f.q[1] = *(const u32x4*)(As + row * LDSK + 16 + half);   // K = 16+half*8+ 0..7
  return f.v;
}
// B 32x16: lane L holds col N=L%16; half=L/16 selects K groups {half*8..} and {16+half*8..} (Bs stored [n][k])
__device__ __forceinline__ v16bf load_frag_b(const unsigned short* Bs, int colBase, int lane) {
  const int col = colBase + (lane & 15);
  const int off = (lane >> 4) << 3;                  // 0 or 8 elements
  union { v16bf v; u32x4 q[2]; } f;
  f.q[0] = *(const u32x4*)(Bs + col * LDSK + off);         // K = half*8   + 0..7
  f.q[1] = *(const u32x4*)(Bs + col * LDSK + off + 16);    // K = 16+half*8+ 0..7
  return f.v;
}

#define WMMA_BF16(a, b, c) \
  __builtin_amdgcn_wmma_f32_16x16x32_bf16(false, (a), false, (b), (short)0, (c), false, false)

// =====================================================================
// Kernel 1: projection GEMM  Y[bt][n] = sum_k X[bt][k] * W[k][n]  (bf16 out)
// =====================================================================
__global__ __launch_bounds__(256) void gemm_proj(const float* __restrict__ X,
                                                 const float* __restrict__ W,
                                                 unsigned short* __restrict__ Y) {
  const int n0 = blockIdx.x * BN;
  const int m0 = blockIdx.y * BM;
  __shared__ __attribute__((aligned(16))) unsigned short As[BM * LDSK];
  __shared__ __attribute__((aligned(16))) unsigned short Bs[BN * LDSK];
  const int tid = threadIdx.x, lane = tid & 31, wave = tid >> 5;
  const int mW = (wave & 3) * 32, nW = (wave >> 2) * 64;
  v8f acc[2][4] = {};

  for (int k0 = 0; k0 < D_; k0 += BK) {
    // prefetch next K-tile into the cache hierarchy (global_prefetch_b8)
    if (k0 + BK < D_) {
      __builtin_prefetch(X + (size_t)(m0 + (tid >> 3)) * D_ + k0 + BK, 0, 1);
      __builtin_prefetch(W + (size_t)(k0 + BK + (tid >> 5)) * D_ + n0 + (tid & 31) * 4, 0, 1);
    }
    { // stage A: X fp32 -> bf16, row-major [m][k]
      int r = tid >> 3; const int c = (tid & 7) * 4;
      #pragma unroll
      for (int p = 0; p < 4; ++p, r += 32) {
        f32x4 v = *(const f32x4*)(X + (size_t)(m0 + r) * D_ + k0 + c);
        #pragma unroll
        for (int i = 0; i < 4; ++i) As[r * LDSK + c + i] = f32_to_bf16(v[i]);
      }
    }
    { // stage B: W fp32, transposed into Bs[n][k]
      int kr = tid >> 5; const int c = (tid & 31) * 4;
      #pragma unroll
      for (int p = 0; p < 4; ++p, kr += 8) {
        f32x4 v = *(const f32x4*)(W + (size_t)(k0 + kr) * D_ + n0 + c);
        #pragma unroll
        for (int i = 0; i < 4; ++i) Bs[(c + i) * LDSK + kr] = f32_to_bf16(v[i]);
      }
    }
    __syncthreads();
    v16bf af[2], bfr[4];
    #pragma unroll
    for (int mi = 0; mi < 2; ++mi) af[mi] = load_frag_a(As, mW + mi * 16, lane);
    #pragma unroll
    for (int ni = 0; ni < 4; ++ni) bfr[ni] = load_frag_b(Bs, nW + ni * 16, lane);
    #pragma unroll
    for (int mi = 0; mi < 2; ++mi)
      #pragma unroll
      for (int ni = 0; ni < 4; ++ni)
        acc[mi][ni] = WMMA_BF16(af[mi], bfr[ni], acc[mi][ni]);
    __syncthreads();
  }
  // epilogue: D layout -> lane L: col N=L%16; VGPR v: row = v + 8*(L/16)
  const int half = lane >> 4, ln = lane & 15;
  #pragma unroll
  for (int mi = 0; mi < 2; ++mi)
    #pragma unroll
    for (int ni = 0; ni < 4; ++ni) {
      const int col = n0 + nW + ni * 16 + ln;
      #pragma unroll
      for (int v = 0; v < 8; ++v) {
        const int row = m0 + mW + mi * 16 + half * 8 + v;
        Y[(size_t)row * D_ + col] = f32_to_bf16(acc[mi][ni][v]);
      }
    }
}

// =====================================================================
// Kernel 2: scores S[b][q][t] = floor((Q[q]·K[t]) / 32), masked t>q -> -inf
// Both tiles are raw bf16 copies -> use async global->LDS (ASYNCcnt) staging.
// =====================================================================
__global__ __launch_bounds__(256) void gemm_scores(const unsigned short* __restrict__ Qb,
                                                   const unsigned short* __restrict__ Kb,
                                                   float* __restrict__ S) {
  const int t0 = blockIdx.x * BN;
  const int q0 = blockIdx.y * BM;
  const int b  = blockIdx.z;
  if (t0 > q0 + (BM - 1)) return;                 // fully masked block
  const unsigned short* Q = Qb + (size_t)b * T_ * D_;
  const unsigned short* K = Kb + (size_t)b * T_ * D_;
  __shared__ __attribute__((aligned(16))) unsigned short As[BM * LDSK];
  __shared__ __attribute__((aligned(16))) unsigned short Bs[BN * LDSK];
  const int tid = threadIdx.x, lane = tid & 31, wave = tid >> 5;
  const int mW = (wave & 3) * 32, nW = (wave >> 2) * 64;
  v8f acc[2][4] = {};

  for (int k0 = 0; k0 < D_; k0 += BK) {
    { // async stage A (Q rows) and B (K rows; K^T means [token][d] == Bs[n][k])
      const int r  = tid >> 2;         // 0..63
      const int cc = (tid & 3) * 8;    // 16-byte chunk within the 32-wide row
      #pragma unroll
      for (int p = 0; p < 2; ++p) {
        const int rr = r + p * 64;
        copy16_g2l(Q + (size_t)(q0 + rr) * D_ + k0 + cc, &As[rr * LDSK + cc]);
        copy16_g2l(K + (size_t)(t0 + rr) * D_ + k0 + cc, &Bs[rr * LDSK + cc]);
      }
      ASYNC_WAIT0();
    }
    __syncthreads();
    v16bf af[2], bfr[4];
    #pragma unroll
    for (int mi = 0; mi < 2; ++mi) af[mi] = load_frag_a(As, mW + mi * 16, lane);
    #pragma unroll
    for (int ni = 0; ni < 4; ++ni) bfr[ni] = load_frag_b(Bs, nW + ni * 16, lane);
    #pragma unroll
    for (int mi = 0; mi < 2; ++mi)
      #pragma unroll
      for (int ni = 0; ni < 4; ++ni)
        acc[mi][ni] = WMMA_BF16(af[mi], bfr[ni], acc[mi][ni]);
    __syncthreads();
  }
  const int half = lane >> 4, ln = lane & 15;
  #pragma unroll
  for (int mi = 0; mi < 2; ++mi)
    #pragma unroll
    for (int ni = 0; ni < 4; ++ni) {
      const int t = t0 + nW + ni * 16 + ln;
      #pragma unroll
      for (int v = 0; v < 8; ++v) {
        const int q = q0 + mW + mi * 16 + half * 8 + v;
        const float val = (t > q) ? -__builtin_inff()
                                  : floorf(acc[mi][ni][v] * 0.03125f); // /sqrt(1024)
        S[((size_t)b * T_ + q) * T_ + t] = val;
      }
    }
}

// =====================================================================
// Kernel 3: column softmax stats (softmax over q axis!): for each (b,t)
// m = max_{q>=t} S, r = 1 / sum_{q>=t} exp(S-m)
// =====================================================================
__global__ __launch_bounds__(256) void col_softmax(const float* __restrict__ S,
                                                   float* __restrict__ cmax,
                                                   float* __restrict__ crcp) {
  const int t = blockIdx.x * 256 + threadIdx.x;
  const int b = blockIdx.y;
  const float* Sb = S + (size_t)b * T_ * T_;
  float m = -__builtin_inff(), s = 0.0f;
  for (int q = t; q < T_; ++q) {                  // coalesced across consecutive t
    const float x = Sb[(size_t)q * T_ + t];
    const float mn = fmaxf(m, x);
    s = s * __expf(m - mn) + __expf(x - mn);
    m = mn;
  }
  cmax[b * T_ + t] = m;
  crcp[b * T_ + t] = 1.0f / s;                    // column always has q=t valid
}

// =====================================================================
// Kernel 4: context C[b][q][d] = sum_{t<=q} softmax(S)[q][t] * V[t][d]
// P is reconstructed in bf16 on the fly while staging the A tile.
// =====================================================================
__global__ __launch_bounds__(256) void gemm_ctx(const float* __restrict__ S,
                                                const float* __restrict__ cmax,
                                                const float* __restrict__ crcp,
                                                const unsigned short* __restrict__ Vb,
                                                float* __restrict__ Out) {
  const int n0 = blockIdx.x * BN;                 // d
  const int q0 = blockIdx.y * BM;                 // q
  const int b  = blockIdx.z;
  const float* Sb = S + (size_t)b * T_ * T_;
  const float* mb = cmax + b * T_;
  const float* rb = crcp + b * T_;
  const unsigned short* V = Vb + (size_t)b * T_ * D_;
  __shared__ __attribute__((aligned(16))) unsigned short As[BM * LDSK];
  __shared__ __attribute__((aligned(16))) unsigned short Bs[BN * LDSK];
  const int tid = threadIdx.x, lane = tid & 31, wave = tid >> 5;
  const int mW = (wave & 3) * 32, nW = (wave >> 2) * 64;
  v8f acc[2][4] = {};

  const int kEnd = q0 + BM;                       // causal: t <= q0+127 only
  for (int k0 = 0; k0 < kEnd; k0 += BK) {
    { // stage A: P = exp(S - m[t]) * rcp[t], 0 above diagonal (never read there)
      int r = tid >> 3; const int c = (tid & 7) * 4;
      #pragma unroll
      for (int p = 0; p < 4; ++p, r += 32) {
        const int q = q0 + r;
        const float* srow = Sb + (size_t)q * T_ + k0 + c;
        #pragma unroll
        for (int i = 0; i < 4; ++i) {
          const int t = k0 + c + i;
          float pv = 0.0f;
          if (t <= q) pv = __expf(srow[i] - mb[t]) * rb[t];
          As[r * LDSK + c + i] = f32_to_bf16(pv);
        }
      }
    }
    { // stage B: V bf16 [t][d] transposed into Bs[d][t]
      int kr = tid >> 5; const int c = (tid & 31) * 4;
      #pragma unroll
      for (int p = 0; p < 4; ++p, kr += 8) {
        u16x4 v = *(const u16x4*)(V + (size_t)(k0 + kr) * D_ + n0 + c);
        #pragma unroll
        for (int i = 0; i < 4; ++i) Bs[(c + i) * LDSK + kr] = v[i];
      }
    }
    __syncthreads();
    v16bf af[2], bfr[4];
    #pragma unroll
    for (int mi = 0; mi < 2; ++mi) af[mi] = load_frag_a(As, mW + mi * 16, lane);
    #pragma unroll
    for (int ni = 0; ni < 4; ++ni) bfr[ni] = load_frag_b(Bs, nW + ni * 16, lane);
    #pragma unroll
    for (int mi = 0; mi < 2; ++mi)
      #pragma unroll
      for (int ni = 0; ni < 4; ++ni)
        acc[mi][ni] = WMMA_BF16(af[mi], bfr[ni], acc[mi][ni]);
    __syncthreads();
  }
  const int half = lane >> 4, ln = lane & 15;
  #pragma unroll
  for (int mi = 0; mi < 2; ++mi)
    #pragma unroll
    for (int ni = 0; ni < 4; ++ni) {
      const int col = n0 + nW + ni * 16 + ln;
      #pragma unroll
      for (int v = 0; v < 8; ++v) {
        const int q = q0 + mW + mi * 16 + half * 8 + v;
        Out[((size_t)b * T_ + q) * D_ + col] = acc[mi][ni][v];
      }
    }
}

// =====================================================================
extern "C" void kernel_launch(void* const* d_in, const int* in_sizes, int n_in,
                              void* d_out, int out_size, void* d_ws, size_t ws_size,
                              hipStream_t stream) {
  const float* X  = (const float*)d_in[0];   // [B,T,D]
  const float* Wq = (const float*)d_in[1];   // [D,D]
  const float* Wk = (const float*)d_in[2];
  const float* Wv = (const float*)d_in[3];
  float* Out = (float*)d_out;                // [B,T,D] fp32

  // Workspace layout: Qb/Kb/Vb bf16 (16 MB each), S fp32 (64 MB), stats (64 KB)
  char* ws = (char*)d_ws;
  unsigned short* Qb = (unsigned short*)(ws);
  unsigned short* Kb = (unsigned short*)(ws + ((size_t)16 << 20));
  unsigned short* Vb = (unsigned short*)(ws + ((size_t)32 << 20));
  float* S  = (float*)(ws + ((size_t)48 << 20));
  float* cm = (float*)(ws + ((size_t)112 << 20));
  float* cr = (float*)(ws + ((size_t)112 << 20) + (size_t)B_ * T_ * sizeof(float));

  const dim3 blk(256);
  const dim3 gProj(D_ / BN, BT_ / BM);       // 8 x 64
  gemm_proj<<<gProj, blk, 0, stream>>>(X, Wq, Qb);
  gemm_proj<<<gProj, blk, 0, stream>>>(X, Wk, Kb);
  gemm_proj<<<gProj, blk, 0, stream>>>(X, Wv, Vb);

  gemm_scores<<<dim3(T_ / BN, T_ / BM, B_), blk, 0, stream>>>(Qb, Kb, S);
  col_softmax<<<dim3(T_ / 256, B_), blk, 0, stream>>>(S, cm, cr);
  gemm_ctx<<<dim3(D_ / BN, T_ / BM, B_), blk, 0, stream>>>(S, cm, cr, Vb, Out);
}
